// QORAttention_90177133347668
// MI455X (gfx1250) — matrix operand order repaired and checked by reading, not compiled, not run on hardware
//
#include <hip/hip_runtime.h>

// ---------------------------------------------------------------------------
// CDNA5 (gfx1250) attention block:
//   rmsnorm -> Q/K/V proj (WMMA bf16) -> RoPE -> flash attention with
//   TDM (tensor_load_to_lds) K/V streaming -> output proj + residual.
// GEMM operands are pre-swizzled into per-lane WMMA fragment order.
// Flash attention: one shared 8KB K+V tile per 32 keys DMA'd by the TDM
// (TENSORcnt), double-buffered in LDS; softmax row-sums computed by WMMA
// against a ones-matrix; row-max via transpose-first lane-local reduction.
// ---------------------------------------------------------------------------

typedef __attribute__((ext_vector_type(16))) __bf16 v16bf;
typedef __attribute__((ext_vector_type(8)))  float  v8f;
typedef __attribute__((ext_vector_type(4)))  unsigned int v4u;
typedef __attribute__((ext_vector_type(4)))  int  v4i;
typedef __attribute__((ext_vector_type(8)))  int  v8i;

#define T_DIM 4096
#define C_DIM 1024
#define H_NUM 16
#define KT    32   // C_DIM / 32 k-tiles per GEMM

__device__ __forceinline__ unsigned short f2bf(float f) {
  unsigned int u = __float_as_uint(f);
  u += 0x7FFFu + ((u >> 16) & 1u);            // round-to-nearest-even
  return (unsigned short)(u >> 16);
}
__device__ __forceinline__ float bf2f(unsigned short h) {
  return __uint_as_float(((unsigned int)h) << 16);
}
__device__ __forceinline__ v16bf load_frag(const unsigned short* p, int lane) {
  return *(const v16bf*)(p + lane * 16);      // 32B contiguous per lane
}
__device__ __forceinline__ v8f wmma_bf16(v16bf a, v16bf b, v8f c) {
  return __builtin_amdgcn_wmma_f32_16x16x32_bf16(false, a, false, b,
                                                 (short)0, c, false, false);
}

// ---------------------------------------------------------------------------
// TDM: DMA a contiguous block (n8 x 8 bytes) of global memory into LDS.
// D# per CDNA5 ISA 8.3/8.4. Tracked by TENSORcnt.
// ---------------------------------------------------------------------------
__device__ __forceinline__ void tdm_load_1d(unsigned int lds_off,
                                            const void* gptr,
                                            unsigned int n8 /*8B units*/) {
  const unsigned long long ga = (unsigned long long)gptr;
  v4u g0;
  g0.x = 1u;                                      // count=1, user descriptor
  g0.y = lds_off;                                 // lds_addr (bytes)
  g0.z = (unsigned int)ga;                        // global_addr[31:0]
  g0.w = ((unsigned int)(ga >> 32) & 0x01FFFFFFu) // global_addr[56:32]
         | (2u << 30);                            // type = 2 ("image")
  v8i g1;
  g1[0] = (int)(3u << 16);                        // data_size = 3 (8 bytes)
  g1[1] = (int)((n8 & 0xFFFFu) << 16);            // tensor_dim0[15:0]
  g1[2] = (int)((n8 >> 16) & 0xFFFFu) | (1 << 16);// tensor_dim0[31:16], td1=1
  g1[3] = (int)((n8 & 0xFFFFu) << 16);            // tile_dim0 = n8
  g1[4] = 1;                                      // tile_dim1 = 1
  g1[5] = (int)n8;                                // tensor_dim0_stride[31:0]
  g1[6] = 0;
  g1[7] = 0;
  v4i g2 = {0, 0, 0, 0}, g3 = {0, 0, 0, 0};
#if defined(__clang_major__) && (__clang_major__ >= 23)
  v8i g4 = {0, 0, 0, 0, 0, 0, 0, 0};
  __builtin_amdgcn_tensor_load_to_lds(g0, g1, g2, g3, g4, 0);
#else
  __builtin_amdgcn_tensor_load_to_lds(g0, g1, g2, g3, 0);
#endif
}

// ---------------------------------------------------------------------------
// RMSNorm + cast to bf16 + scatter into A-fragment order.
// A-layout element (m, k): lane = m + 16*((k/8)&1), slot = (k/16)*8 + (k&7)
// ---------------------------------------------------------------------------
__global__ void rmsnorm_swz_kernel(const float* __restrict__ x,
                                   const float* __restrict__ nw,
                                   unsigned short* __restrict__ xnA) {
  const int t = blockIdx.x, tid = threadIdx.x;
  const float4 v = ((const float4*)(x + (size_t)t * C_DIM))[tid];
  float vals[4] = {v.x, v.y, v.z, v.w};
  float ss = vals[0]*vals[0] + vals[1]*vals[1] + vals[2]*vals[2] + vals[3]*vals[3];
#pragma unroll
  for (int m = 1; m < 32; m <<= 1) ss += __shfl_xor(ss, m, 32);
  __shared__ float red[8];
  if ((tid & 31) == 0) red[tid >> 5] = ss;
  __syncthreads();
  float tot = 0.f;
#pragma unroll
  for (int i = 0; i < 8; ++i) tot += red[i];
  const float inv = rsqrtf(tot * (1.0f / 1024.0f) + 1e-6f);
#pragma unroll
  for (int e = 0; e < 4; ++e) {
    const int c = tid * 4 + e;
    const float o = vals[e] * inv * nw[c];
    const int m = t & 15, k = c & 31, g = k >> 3;
    const int lane = m + 16 * (g & 1), slot = (g >> 1) * 8 + (k & 7);
    xnA[(((size_t)(t >> 4)) * KT + (c >> 5)) * 512 + lane * 16 + slot] = f2bf(o);
  }
}

// Weight -> bf16 B-fragment order: B[k][n] = W[n0+n][k0+k].
// B-layout: lane = n + 16*((k/8)&1), slot = (k/16)*8 + (k&7).
__global__ void wswzB_kernel(const float* __restrict__ W,
                             unsigned short* __restrict__ WB) {
  const size_t stride = (size_t)gridDim.x * blockDim.x;
  for (size_t idx = (size_t)blockIdx.x * blockDim.x + threadIdx.x;
       idx < (size_t)C_DIM * C_DIM; idx += stride) {
    const int d = (int)(idx >> 10), c = (int)(idx & 1023);
    const int n = d & 15, k = c & 31;
    const int lane = n + 16 * ((k >> 3) & 1), slot = (k >> 4) * 8 + (k & 7);
    WB[(((size_t)(d >> 4)) * KT + (c >> 5)) * 512 + lane * 16 + slot] = f2bf(W[idx]);
  }
}

// ---------------------------------------------------------------------------
// GEMM: M=4096, N=1024, K=1024, fragment-order operands, L2-resident reuse.
// Block 256 = 8 waves, block tile 128x128, wave tile 64x32 (4x2 WMMA tiles).
// EPI 0: bf16 [T][C] output.  EPI 1: f32 output = residual + acc.
// ---------------------------------------------------------------------------
template <int EPI>
__global__ void gemm_wmma_kernel(const unsigned short* __restrict__ Af,
                                 const unsigned short* __restrict__ Bf,
                                 unsigned short* __restrict__ outB,
                                 float* __restrict__ outF,
                                 const float* __restrict__ resid) {
  const int lane = threadIdx.x & 31, wave = threadIdx.x >> 5;
  const int waveM0 = blockIdx.y * 128 + (wave & 1) * 64;
  const int waveN0 = blockIdx.x * 128 + (wave >> 1) * 32;
  const int mT = waveM0 >> 4, nT = waveN0 >> 4;
  const v8f vz = {0.f, 0.f, 0.f, 0.f, 0.f, 0.f, 0.f, 0.f};
  v8f acc[4][2];
#pragma unroll
  for (int i = 0; i < 4; ++i)
#pragma unroll
    for (int j = 0; j < 2; ++j) acc[i][j] = vz;

  for (int kt = 0; kt < KT; ++kt) {
    if (kt + 1 < KT) {  // global_prefetch_b8 next K-tile
      __builtin_prefetch(Af + ((size_t)mT * KT + kt + 1) * 512 + lane * 16, 0, 0);
      __builtin_prefetch(Bf + ((size_t)nT * KT + kt + 1) * 512 + lane * 16, 0, 0);
    }
    v16bf a[4], b[2];
#pragma unroll
    for (int i = 0; i < 4; ++i)
      a[i] = load_frag(Af + ((size_t)(mT + i) * KT + kt) * 512, lane);
#pragma unroll
    for (int j = 0; j < 2; ++j)
      b[j] = load_frag(Bf + ((size_t)(nT + j) * KT + kt) * 512, lane);
#pragma unroll
    for (int i = 0; i < 4; ++i)
#pragma unroll
      for (int j = 0; j < 2; ++j)
        acc[i][j] = wmma_bf16(a[i], b[j], acc[i][j]);
  }

  const int ln = lane & 15, lh = lane >> 4;
#pragma unroll
  for (int i = 0; i < 4; ++i)
#pragma unroll
    for (int j = 0; j < 2; ++j)
#pragma unroll
      for (int r = 0; r < 8; ++r) {
        const size_t idx =
            (size_t)(waveM0 + i * 16 + r + 8 * lh) * C_DIM + (waveN0 + j * 16 + ln);
        const float vv = acc[i][j][r];
        if (EPI == 0) outB[idx] = f2bf(vv);
        else          outF[idx] = resid[idx] + vv;
      }
}

// ---------------------------------------------------------------------------
// RoPE on Q,K (1/sqrt(64) folded into Q) + scatter into fragment orders.
// QA : A-frags per (head, 16-query tile, d-chunk of 32)
// KVB: combined per (head, 32-key step j): 4KB K B-frags + 4KB V B-frags
//      -> one 8KB contiguous TDM tile per j.
// ---------------------------------------------------------------------------
__global__ void attn_prep_kernel(const unsigned short* __restrict__ Qp,
                                 const unsigned short* __restrict__ Kp,
                                 const unsigned short* __restrict__ Vp,
                                 unsigned short* __restrict__ QA,
                                 unsigned short* __restrict__ KVB) {
  const int t = blockIdx.x, tid = threadIdx.x;
  for (int it = tid; it < 512; it += 256) {      // (head, rot-pair)
    const int h = it >> 5, dp = it & 31;
    const float freq = __powf(10000.0f, -(float)dp * (1.0f / 32.0f));
    float sn, cs;
    __sincosf((float)t * freq, &sn, &cs);
    const size_t base = (size_t)t * C_DIM + h * 64 + dp;
    const float q1 = bf2f(Qp[base]), q2 = bf2f(Qp[base + 32]);
    const float k1 = bf2f(Kp[base]), k2 = bf2f(Kp[base + 32]);
    const float qo1 = (q1 * cs - q2 * sn) * 0.125f;
    const float qo2 = (q2 * cs + q1 * sn) * 0.125f;
    const float ko1 = k1 * cs - k2 * sn;
    const float ko2 = k2 * cs + k1 * sn;
    const size_t kvTile = ((size_t)h * (T_DIM >> 5) + (t >> 5)) * 4096;
    auto wq = [&](int kk, float v) {  // A-layout, m = t&15
      const int kc = kk >> 5, k = kk & 31, g = k >> 3;
      const int lane = (t & 15) + 16 * (g & 1), slot = (g >> 1) * 8 + (k & 7);
      QA[(((size_t)h * (T_DIM >> 4) + (t >> 4)) * 2 + kc) * 512 + lane * 16 + slot] =
          f2bf(v);
    };
    auto wk = [&](int kk, float v) {  // B-layout, n = t&15 (key in 16-tile)
      const int kc = kk >> 5, k = kk & 31;
      const int lane = (t & 15) + 16 * ((k >> 3) & 1), slot = (k >> 4) * 8 + (k & 7);
      KVB[kvTile + ((((t >> 4) & 1) * 2 + kc)) * 512 + lane * 16 + slot] = f2bf(v);
    };
    wq(dp, qo1); wq(dp + 32, qo2);
    wk(dp, ko1); wk(dp + 32, ko2);
  }
  for (int it = tid; it < 1024; it += 256) {     // V: (head, d)
    const int h = it >> 6, d = it & 63;
    const unsigned short v = Vp[(size_t)t * C_DIM + h * 64 + d];
    const int k = t & 31, n = d & 15, nc = d >> 4;
    const int lane = n + 16 * ((k >> 3) & 1), slot = (k >> 4) * 8 + (k & 7);
    KVB[((size_t)h * (T_DIM >> 5) + (t >> 5)) * 4096 + 2048 + nc * 512 +
        lane * 16 + slot] = v;
  }
}

// ---------------------------------------------------------------------------
// Flash attention: block = 1 head x 128 queries (8 waves x 16-query tiles).
// Wave 0 DMAs one 8KB K+V tile per 32 keys via TDM, double-buffered in LDS.
// Softmax: transpose-first (S -> LDS f32 -> A-layout), lane-local max + one
// shfl_xor(16); row sums & l-update fused into one WMMA vs a ones-matrix.
// Output: normalized O goes through LDS and is written as coalesced A-frags.
// ---------------------------------------------------------------------------
__global__ void flash_attn_kernel(const unsigned short* __restrict__ QA,
                                  const unsigned short* __restrict__ KVB,
                                  unsigned short* __restrict__ AOA) {
  __shared__ unsigned short kv[2][4096];   // 2 slots x 8KB (4KB K | 4KB V)
  __shared__ float pS[8][1024];            // per-wave transpose tile (4KB)
  const int lane = threadIdx.x & 31, wave = threadIdx.x >> 5;
  const int h = blockIdx.y;
  const int t0 = blockIdx.x * 128 + wave * 16;
  const int ln = lane & 15, lh = lane >> 4;
  const v8f vz = {0.f, 0.f, 0.f, 0.f, 0.f, 0.f, 0.f, 0.f};

  const unsigned short* qb = QA + ((size_t)h * (T_DIM >> 4) + (t0 >> 4)) * 2 * 512;
  const v16bf q0 = load_frag(qb, lane);
  const v16bf q1 = load_frag(qb + 512, lane);

  union { unsigned int u[8]; v16bf v; } ones;  // B-frag of all 1.0bf16
#pragma unroll
  for (int i = 0; i < 8; ++i) ones.u[i] = 0x3F803F80u;

  v8f O[4];
#pragma unroll
  for (int c4 = 0; c4 < 4; ++c4) O[c4] = vz;
  v8f lv = vz;                 // per-row softmax denominator (C-layout)
  float m_lane = -3.0e38f;     // running max of row (lane & 15)

  const unsigned short* kvsrc = KVB + (size_t)h * (T_DIM >> 5) * 4096;
  const unsigned int kvLds = (unsigned int)(unsigned long long)(const void*)&kv[0][0];
  const int jEnd = (blockIdx.x * 128 + 112 + 16 + 31) >> 5;  // block-uniform

  if (wave == 0) tdm_load_1d(kvLds, kvsrc, 1024);            // tile j=0

  float* pw = pS[wave];
  for (int j = 0; j < jEnd; ++j) {
    const int slot = j & 1;
    if (wave == 0) {
      if (j + 1 < jEnd) {
        tdm_load_1d(kvLds + (1 - slot) * 8192, kvsrc + (size_t)(j + 1) * 4096, 1024);
        __builtin_amdgcn_s_wait_tensorcnt(1);  // tile j done, j+1 in flight
      } else {
        __builtin_amdgcn_s_wait_tensorcnt(0);
      }
    }
    __syncthreads();                            // tile j visible to all waves
    const unsigned short* kt = kv[slot];

    v8f S[2];
#pragma unroll
    for (int nc = 0; nc < 2; ++nc) {
      v8f s = vz;
      s = wmma_bf16(q0, load_frag(kt + (nc * 2 + 0) * 512, lane), s);
      s = wmma_bf16(q1, load_frag(kt + (nc * 2 + 1) * 512, lane), s);
      S[nc] = s;
    }

    // S (C-layout, f32) -> LDS [16 rows][32 keys]
#pragma unroll
    for (int nc = 0; nc < 2; ++nc)
#pragma unroll
      for (int r = 0; r < 8; ++r)
        pw[(r + 8 * lh) * 32 + nc * 16 + ln] = S[nc][r];
    asm volatile("s_wait_dscnt 0" ::: "memory");

    // Re-read in A-layout: lane holds 16 entries of row (t0 + ln)
    float sv[16];
#pragma unroll
    for (int jj = 0; jj < 8; ++jj) {
      const int k = (jj < 4 ? 2 * jj : 16 + 2 * (jj - 4)) + (lh ? 8 : 0);
      sv[2 * jj]     = pw[ln * 32 + k];
      sv[2 * jj + 1] = pw[ln * 32 + k + 1];
    }
    if (j * 32 + 31 > t0) {                     // wave-uniform: causal mask
#pragma unroll
      for (int jj = 0; jj < 8; ++jj) {
        const int k = (jj < 4 ? 2 * jj : 16 + 2 * (jj - 4)) + (lh ? 8 : 0);
        if (j * 32 + k     > t0 + ln) sv[2 * jj]     = -3.0e38f;
        if (j * 32 + k + 1 > t0 + ln) sv[2 * jj + 1] = -3.0e38f;
      }
    }
    float mx = sv[0];
#pragma unroll
    for (int i = 1; i < 16; ++i) mx = fmaxf(mx, sv[i]);
    mx = fmaxf(mx, __shfl_xor(mx, 16, 32));     // merge the two k-halves
    const float nm = fmaxf(m_lane, mx);
    const float rs_lane = __expf(m_lane - nm);
    m_lane = nm;

    union { unsigned int u[8]; v16bf v; } pf;   // P in A-frag order
#pragma unroll
    for (int jj = 0; jj < 8; ++jj) {
      const float p0 = __expf(sv[2 * jj]     - nm);
      const float p1 = __expf(sv[2 * jj + 1] - nm);
      pf.u[jj] = (unsigned int)f2bf(p0) | ((unsigned int)f2bf(p1) << 16);
    }

    // Broadcast rescale factor to C-layout rows r + 8*lh
    float rsC[8];
#pragma unroll
    for (int r = 0; r < 8; ++r) rsC[r] = __shfl(rs_lane, r + 8 * lh, 32);
#pragma unroll
    for (int c4 = 0; c4 < 4; ++c4)
#pragma unroll
      for (int r = 0; r < 8; ++r) O[c4][r] *= rsC[r];
#pragma unroll
    for (int r = 0; r < 8; ++r) lv[r] *= rsC[r];

    lv = wmma_bf16(pf.v, ones.v, lv);           // l += P x 1 (row sums)

    const unsigned short* vt = kv[slot] + 2048;
#pragma unroll
    for (int c4 = 0; c4 < 4; ++c4)
      O[c4] = wmma_bf16(pf.v, load_frag(vt + c4 * 512, lane), O[c4]);

    __syncthreads();                            // all done reading slot j
  }

  // Normalize, transpose through LDS, emit two coalesced A-frags per wave.
  float invl[8];
#pragma unroll
  for (int r = 0; r < 8; ++r) invl[r] = 1.0f / lv[r];
#pragma unroll
  for (int c4 = 0; c4 < 4; ++c4)
#pragma unroll
    for (int r = 0; r < 8; ++r)
      pw[(r + 8 * lh) * 64 + c4 * 16 + ln] = O[c4][r] * invl[r];
  asm volatile("s_wait_dscnt 0" ::: "memory");

#pragma unroll
  for (int kc = 0; kc < 2; ++kc) {
    union { unsigned int u[8]; v16bf v; } of;
#pragma unroll
    for (int jj = 0; jj < 8; ++jj) {
      const int k = (jj < 4 ? 2 * jj : 16 + 2 * (jj - 4)) + (lh ? 8 : 0);
      const float o0 = pw[ln * 64 + kc * 32 + k];
      const float o1 = pw[ln * 64 + kc * 32 + k + 1];
      of.u[jj] = (unsigned int)f2bf(o0) | ((unsigned int)f2bf(o1) << 16);
    }
    unsigned short* dst =
        AOA + (((size_t)(t0 >> 4)) * KT + (h * 2 + kc)) * 512 + lane * 16;
    *(v16bf*)dst = of.v;
  }
}

// ---------------------------------------------------------------------------
extern "C" void kernel_launch(void* const* d_in, const int* in_sizes, int n_in,
                              void* d_out, int out_size, void* d_ws, size_t ws_size,
                              hipStream_t stream) {
  const float* x  = (const float*)d_in[0];
  const float* wq = (const float*)d_in[1];
  const float* wk = (const float*)d_in[2];
  const float* wv = (const float*)d_in[3];
  const float* wo = (const float*)d_in[4];
  const float* nw = (const float*)d_in[5];
  float* out = (float*)d_out;

  char* ws = (char*)d_ws;
  const size_t MB = (size_t)1 << 20;
  unsigned short* xnA = (unsigned short*)(ws + 0 * MB);   // 8 MB A-frags
  unsigned short* WqB = (unsigned short*)(ws + 8 * MB);   // 2 MB B-frags
  unsigned short* WkB = (unsigned short*)(ws + 10 * MB);
  unsigned short* WvB = (unsigned short*)(ws + 12 * MB);
  unsigned short* WoB = (unsigned short*)(ws + 14 * MB);
  unsigned short* Qp  = (unsigned short*)(ws + 16 * MB);  // 8 MB plain bf16
  unsigned short* Kp  = (unsigned short*)(ws + 24 * MB);
  unsigned short* Vp  = (unsigned short*)(ws + 32 * MB);
  unsigned short* QA  = (unsigned short*)(ws + 40 * MB);  // roped Q A-frags
  unsigned short* KVB = (unsigned short*)(ws + 48 * MB);  // 16 MB K+V TDM tiles
  unsigned short* AOA = (unsigned short*)(ws + 64 * MB);  // attn out A-frags

  rmsnorm_swz_kernel<<<T_DIM, 256, 0, stream>>>(x, nw, xnA);
  wswzB_kernel<<<1024, 256, 0, stream>>>(wq, WqB);
  wswzB_kernel<<<1024, 256, 0, stream>>>(wk, WkB);
  wswzB_kernel<<<1024, 256, 0, stream>>>(wv, WvB);
  wswzB_kernel<<<1024, 256, 0, stream>>>(wo, WoB);

  dim3 gGrid(C_DIM / 128, T_DIM / 128);
  gemm_wmma_kernel<0><<<gGrid, 256, 0, stream>>>(xnA, WqB, Qp, nullptr, nullptr);
  gemm_wmma_kernel<0><<<gGrid, 256, 0, stream>>>(xnA, WkB, Kp, nullptr, nullptr);
  gemm_wmma_kernel<0><<<gGrid, 256, 0, stream>>>(xnA, WvB, Vp, nullptr, nullptr);

  attn_prep_kernel<<<T_DIM, 256, 0, stream>>>(Qp, Kp, Vp, QA, KVB);
  flash_attn_kernel<<<dim3(T_DIM / 128, H_NUM), 256, 0, stream>>>(QA, KVB, AOA);

  gemm_wmma_kernel<1><<<gGrid, 256, 0, stream>>>(AOA, WoB, nullptr, out, x);
}
